// SelfAttention_23072564314471
// MI455X (gfx1250) — matrix operand order repaired and checked by reading, not compiled, not run on hardware
//
#include <hip/hip_runtime.h>
#include <hip/hip_bf16.h>
#include <cmath>
#include <cstdint>

// ---------------------------------------------------------------------------
// Problem constants (match reference setup_inputs)
// ---------------------------------------------------------------------------
#define BATCH   2
#define SEQ     2048
#define CDIM    1024
#define NHEAD   16
#define HDIM    64
#define BH      (BATCH * NHEAD)          // 32
#define MROWS   (BATCH * SEQ)            // 4096
#define NQKV    (3 * CDIM)               // 3072
#define EPSY    1e-6f

typedef __attribute__((ext_vector_type(16))) __bf16   v16bf;
typedef __attribute__((ext_vector_type(8)))  __bf16   v8bf;
typedef __attribute__((ext_vector_type(8)))  float    v8f;
typedef __attribute__((ext_vector_type(4)))  uint32_t v4u;
typedef __attribute__((ext_vector_type(8)))  uint32_t v8u;

// ---------------------------------------------------------------------------
// WMMA helper: D = A(16x32 bf16) * B(32x16 bf16) + C(16x16 f32)
// ---------------------------------------------------------------------------
__device__ __forceinline__ v8f wmma_bf16(v16bf a, v16bf b, v8f c) {
  return __builtin_amdgcn_wmma_f32_16x16x32_bf16(
      /*neg_a=*/false, a, /*neg_b=*/false, b,
      /*c_mod=*/(short)0, c, /*reuse_a=*/false, /*reuse_b=*/false);
}

// Assemble a v16bf from two 16-byte (8 x bf16) chunks.
__device__ __forceinline__ v16bf load16_2(const __bf16* p0, const __bf16* p1) {
  v8bf lo = *(const v8bf*)p0;
  v8bf hi = *(const v8bf*)p1;
  v16bf r;
#pragma unroll
  for (int i = 0; i < 8; ++i) { r[i] = lo[i]; r[i + 8] = hi[i]; }
  return r;
}
// 16 contiguous bf16
__device__ __forceinline__ v16bf load16(const __bf16* p) { return load16_2(p, p + 8); }

// ---------------------------------------------------------------------------
// CDNA5 async copy: global -> LDS, 16B per lane, tracked by ASYNCcnt
// ---------------------------------------------------------------------------
__device__ __forceinline__ void async_copy16_to_lds(uint32_t lds_byte_addr,
                                                    const void* gptr) {
  uint64_t ga = (uint64_t)(uintptr_t)gptr;
  asm volatile("global_load_async_to_lds_b128 %0, %1, off"
               :: "v"(lds_byte_addr), "v"(ga) : "memory");
}
__device__ __forceinline__ void wait_async_all() {
  asm volatile("s_wait_asynccnt 0" ::: "memory");
}

// ---------------------------------------------------------------------------
// CDNA5 Tensor Data Mover: one descriptor moves a contiguous panel
// (n8 x 8 bytes) from global memory into LDS.  Tracked by TENSORcnt.
// D# layout per CDNA5 ISA.
// ---------------------------------------------------------------------------
__device__ __forceinline__ void tdm_load_panel_to_lds(const void* gsrc,
                                                      uint32_t lds_byte_addr,
                                                      uint32_t n8 /* 8B units */) {
  const uint64_t ga = (uint64_t)(uintptr_t)gsrc;
  v4u g0;
  g0[0] = 1u;                                              // count=1 (valid user D#)
  g0[1] = lds_byte_addr;                                   // lds_addr [63:32]
  g0[2] = (uint32_t)ga;                                    // global_addr [95:64]
  g0[3] = (uint32_t)((ga >> 32) & 0x1FFFFFFu) | (2u << 30);// addr hi + type=2 (image)
  v8u g1;
  g1[0] = 3u << 16;                 // workgroup_mask=0, data_size=3 (8B)
  g1[1] = (n8 & 0xFFFFu) << 16;     // tensor_dim0[15:0]  @ bits 63:48
  g1[2] = ((n8 >> 16) & 0xFFFFu)    // tensor_dim0[31:16] @ bits 79:64
        | (1u << 16);               // tensor_dim1 = 1    @ bits 111:80 (lo16)
  g1[3] = (n8 & 0xFFFFu) << 16;     // tile_dim0          @ bits 127:112
  g1[4] = 1u;                       // tile_dim1=1, tile_dim2=0
  g1[5] = n8;                       // tensor_dim0_stride[31:0]
  g1[6] = 0u;                       // stride0 hi, stride1 lo
  g1[7] = 0u;
  asm volatile("tensor_load_to_lds %0, %1" :: "s"(g0), "s"(g1) : "memory");
}

// ---------------------------------------------------------------------------
// Prep 1: fp32 rows -> bf16 rows + per-row sum of squares.  grid = rows
// ---------------------------------------------------------------------------
__global__ void rows_to_bf16_sumsq(const float* __restrict__ src,
                                   __bf16* __restrict__ dst,
                                   float* __restrict__ sumsq, int K) {
  const int row = blockIdx.x;
  const float* s = src + (size_t)row * K;
  __bf16* d = dst + (size_t)row * K;
  float acc = 0.f;
  for (int k = threadIdx.x; k < K; k += blockDim.x) {
    float v = s[k];
    acc += v * v;
    d[k] = (__bf16)v;
  }
  __shared__ float red[256];
  red[threadIdx.x] = acc;
  __syncthreads();
  for (int s2 = 128; s2 > 0; s2 >>= 1) {
    if (threadIdx.x < s2) red[threadIdx.x] += red[threadIdx.x + s2];
    __syncthreads();
  }
  if (threadIdx.x == 0) sumsq[row] = red[0];
}

// ---------------------------------------------------------------------------
// Prep 2: tiled transpose w[K][N] fp32 -> wT[N][K] bf16, plus per-column
// sum-of-squares via 32-lane butterfly + global_atomic_add_f32.
// block = (32, 8), each block moves a 32x32 tile.  wsq must be pre-zeroed.
// ---------------------------------------------------------------------------
__global__ void __launch_bounds__(256)
transpose_bf16_colsumsq(const float* __restrict__ w, __bf16* __restrict__ wT,
                        float* __restrict__ wsq, int K, int N) {
  __shared__ float tile[32][33];
  const int tx = threadIdx.x;          // 0..31
  const int ty = threadIdx.y;          // 0..7
  const int n0 = blockIdx.x * 32;
  const int k0 = blockIdx.y * 32;

  // coalesced loads: row k0+ty+8i, cols n0+tx
#pragma unroll
  for (int i = 0; i < 4; ++i) {
    tile[ty + 8 * i][tx] = w[(size_t)(k0 + ty + 8 * i) * N + n0 + tx];
  }
  __syncthreads();

  // coalesced stores: wT[n0+nl][k0+tx] = tile[tx][nl]; butterfly-sum squares
#pragma unroll
  for (int i = 0; i < 4; ++i) {
    const int nl = ty + 8 * i;
    const float v = tile[tx][nl];
    wT[(size_t)(n0 + nl) * K + k0 + tx] = (__bf16)v;
    float s = v * v;
#pragma unroll
    for (int msk = 1; msk < 32; msk <<= 1) s += __shfl_xor(s, msk, 32);
    if (tx == 0) atomicAdd(&wsq[n0 + nl], s);
  }
}

// ---------------------------------------------------------------------------
// Wave-level GEMM core, double-buffered with independent tile buffers
// (ping-pong; no rotation copies):
//   acc[4] (16x64 strip) += A[16 x 1024] (in LDS) * B[1024 x 64]
// ldsA: 16 rows x CDIM bf16.  BT column-major (BT[n][k]).
// ---------------------------------------------------------------------------
struct KTile {
  v16bf a;
  v16bf b[4];
};

__device__ __forceinline__ void load_tile(KTile& t, const __bf16* __restrict__ Arow,
                                          const __bf16* const Brow[4], int k, int g) {
  t.a = load16(Arow + k + 16 * g);
#pragma unroll
  for (int j = 0; j < 4; ++j) t.b[j] = load16(Brow[j] + k);
}

__device__ __forceinline__ void mma_tile(const KTile& t, v8f acc[4]) {
#pragma unroll
  for (int j = 0; j < 4; ++j) acc[j] = wmma_bf16(t.a, t.b[j], acc[j]);
}

__device__ __forceinline__ void wave_gemm_16x64_lds(const __bf16* __restrict__ ldsA,
                                                    const __bf16* __restrict__ BT,
                                                    int nbase, int g, int ln,
                                                    v8f acc[4]) {
  const __bf16* Arow = ldsA + (size_t)ln * CDIM;
  const __bf16* Brow[4];
#pragma unroll
  for (int j = 0; j < 4; ++j) {
    Brow[j] = BT + (size_t)(nbase + 16 * j + ln) * CDIM + 16 * g;
    __builtin_prefetch(Brow[j], 0, 1);           // global_prefetch_b8
  }

  KTile t0, t1;
  load_tile(t0, Arow, Brow, 0, g);
#pragma unroll 1
  for (int k = 0; k < CDIM - 64; k += 64) {
#pragma unroll
    for (int j = 0; j < 4; ++j) __builtin_prefetch(Brow[j] + k + 512, 0, 1);
    load_tile(t1, Arow, Brow, k + 32, g);
    mma_tile(t0, acc);
    load_tile(t0, Arow, Brow, k + 64, g);
    mma_tile(t1, acc);
  }
  load_tile(t1, Arow, Brow, CDIM - 32, g);
  mma_tile(t0, acc);
  mma_tile(t1, acc);
}

// Stage the contiguous 16-row A panel (16*CDIM bf16 = 32 KB) into LDS.
// Variant 1: per-lane CDNA5 async loads (ASYNCcnt).
__device__ __forceinline__ void stage_A_panel_async(const __bf16* __restrict__ Apanel,
                                                    __bf16* __restrict__ ldsA) {
  const uint32_t lds_base = (uint32_t)(uintptr_t)ldsA;
  const uint32_t tid16 = threadIdx.x * 16;
#pragma unroll
  for (int i = 0; i < 8; ++i) {
    const uint32_t off = (uint32_t)i * 4096u + tid16;
    async_copy16_to_lds(lds_base + off, (const char*)Apanel + off);
  }
  wait_async_all();
  __syncthreads();
}

// Variant 2: one Tensor Data Mover descriptor issued by wave 0 (TENSORcnt).
__device__ __forceinline__ void stage_A_panel_tdm(const __bf16* __restrict__ Apanel,
                                                  __bf16* __restrict__ ldsA) {
  if (threadIdx.x < 32) {
    tdm_load_panel_to_lds(Apanel, (uint32_t)(uintptr_t)ldsA,
                          (16u * CDIM * 2u) / 8u);   // 4096 x 8B
    __builtin_amdgcn_s_wait_tensorcnt(0);
  }
  __syncthreads();
}

__device__ __forceinline__ float yat_val(float dot, float xs, float ws,
                                         float scale, float bias) {
  float dist = xs + ws - 2.f * dot + EPSY;
  dist = fmaxf(dist, EPSY);
  return dot * dot / dist * scale + bias;
}

// ---------------------------------------------------------------------------
// YAT QKV GEMM: [4096 x 1024] x [1024 x 3072], scatter q/k -> [BH,T,64],
// v -> transposed [BH,64,T].  block = 256 (8 waves), block tile = 16 x 512.
// A panel staged via Tensor Data Mover.
// ---------------------------------------------------------------------------
__global__ void __launch_bounds__(256)
yat_qkv_kernel(const __bf16* __restrict__ Ab, const __bf16* __restrict__ BT,
               const float* __restrict__ xsq, const float* __restrict__ wsq,
               const float* __restrict__ bias,
               __bf16* __restrict__ qb, __bf16* __restrict__ kb,
               __bf16* __restrict__ vT, float scale) {
  const int wave = threadIdx.x >> 5;
  const int lane = threadIdx.x & 31;
  const int g = lane >> 4, ln = lane & 15;
  const int mbase = blockIdx.x * 16;
  const int nbase = blockIdx.y * 512 + wave * 64;

  __shared__ __attribute__((aligned(16))) __bf16 ldsA[16 * CDIM];  // 32 KB
  stage_A_panel_tdm(Ab + (size_t)mbase * CDIM, ldsA);

  v8f acc[4] = {};
  wave_gemm_16x64_lds(ldsA, BT, nbase, g, ln, acc);

#pragma unroll
  for (int j = 0; j < 4; ++j) {
    const int col = nbase + 16 * j + ln;
    const float ws = wsq[col];
    const float bs = bias[col];
    const int sel = col >> 10;        // 0=q 1=k 2=v
    const int cc = col & 1023;
    const int h = cc >> 6;
    const int d = cc & 63;
#pragma unroll
    for (int r = 0; r < 8; ++r) {
      const int row = mbase + r + 8 * g;       // C-layout: lanes<16 -> M=r, else M=8+r
      const float y = yat_val(acc[j][r], xsq[row], ws, scale, bs);
      const int b = row >> 11;                 // / SEQ
      const int t = row & (SEQ - 1);
      const int bhid = b * NHEAD + h;
      if (sel == 0)
        qb[((size_t)bhid * SEQ + t) * HDIM + d] = (__bf16)y;
      else if (sel == 1)
        kb[((size_t)bhid * SEQ + t) * HDIM + d] = (__bf16)y;
      else
        vT[((size_t)bhid * HDIM + d) * SEQ + t] = (__bf16)y;
    }
  }
}

// ---------------------------------------------------------------------------
// YAT projection GEMM: [4096 x 1024] x [1024 x 1024] -> f32 d_out
// A panel staged via per-lane async loads.
// ---------------------------------------------------------------------------
__global__ void __launch_bounds__(256)
yat_proj_kernel(const __bf16* __restrict__ Ab, const __bf16* __restrict__ BT,
                const float* __restrict__ xsq, const float* __restrict__ wsq,
                const float* __restrict__ bias, float* __restrict__ out,
                float scale) {
  const int wave = threadIdx.x >> 5;
  const int lane = threadIdx.x & 31;
  const int g = lane >> 4, ln = lane & 15;
  const int mbase = blockIdx.x * 16;
  const int nbase = blockIdx.y * 512 + wave * 64;

  __shared__ __attribute__((aligned(16))) __bf16 ldsA[16 * CDIM];  // 32 KB
  stage_A_panel_async(Ab + (size_t)mbase * CDIM, ldsA);

  v8f acc[4] = {};
  wave_gemm_16x64_lds(ldsA, BT, nbase, g, ln, acc);

#pragma unroll
  for (int j = 0; j < 4; ++j) {
    const int col = nbase + 16 * j + ln;
    const float ws = wsq[col];
    const float bs = bias[col];
#pragma unroll
    for (int r = 0; r < 8; ++r) {
      const int row = mbase + r + 8 * g;
      out[(size_t)row * CDIM + col] = yat_val(acc[j][r], xsq[row], ws, scale, bs);
    }
  }
}

// ---------------------------------------------------------------------------
// RoPE in-place on q,k bf16 [BH, T, 64]; one thread per (bh, t, pair)
// ---------------------------------------------------------------------------
__global__ void rope_kernel(__bf16* __restrict__ qb, __bf16* __restrict__ kb) {
  const int tid = blockIdx.x * blockDim.x + threadIdx.x;
  const int total = BH * SEQ * (HDIM / 2);
  if (tid >= total) return;
  const int i = tid & 31;              // frequency index, half = 32
  const int t = (tid >> 5) & (SEQ - 1);
  const int bhid = tid >> 16;          // / (32 * 2048)
  // freq = 10000^(-i/32) = exp(-i * ln(10000)/32)
  const float freq = __expf(-(float)i * 0.2878231366f);
  const float ang = (float)t * freq;
  float s, c;
  __sincosf(ang, &s, &c);
  const size_t base = ((size_t)bhid * SEQ + t) * HDIM + 2 * i;
  {
    float t0 = (float)qb[base], t1 = (float)qb[base + 1];
    qb[base]     = (__bf16)(t0 * c - t1 * s);
    qb[base + 1] = (__bf16)(t1 * c + t0 * s);
  }
  {
    float t0 = (float)kb[base], t1 = (float)kb[base + 1];
    kb[base]     = (__bf16)(t0 * c - t1 * s);
    kb[base + 1] = (__bf16)(t1 * c + t0 * s);
  }
}

// ---------------------------------------------------------------------------
// Causal flash attention: 1 wave (32 threads) per (bh, 16-query tile)
// q,k: [BH,T,64] bf16; vT: [BH,64,T] bf16; Of: [B,T,C] f32
// ---------------------------------------------------------------------------
__global__ void __launch_bounds__(32)
attn_kernel(const __bf16* __restrict__ qb, const __bf16* __restrict__ kb,
            const __bf16* __restrict__ vT, float* __restrict__ Of) {
  const int NQ = SEQ / 16;                  // 128 query tiles
  const int bhid = blockIdx.x / NQ;
  const int qidx = blockIdx.x % NQ;
  const int q0 = qidx * 16;
  const int lane = threadIdx.x;
  const int g = lane >> 4, ln = lane & 15;

  const __bf16* Q = qb + (size_t)bhid * SEQ * HDIM;
  const __bf16* Kp = kb + (size_t)bhid * SEQ * HDIM;
  const __bf16* Vp = vT + (size_t)bhid * HDIM * SEQ;

  // Q A-tiles (depth 0..31 and 32..63), loaded once
  const __bf16* Qrow = Q + (size_t)(q0 + ln) * HDIM;
  const v16bf a0 = load16(Qrow + 16 * g);
  const v16bf a1 = load16(Qrow + 32 + 16 * g);

  v8f o[4] = {};
  float mrow[8], lrow[8];
#pragma unroll
  for (int r = 0; r < 8; ++r) { mrow[r] = -3.0e38f; lrow[r] = 0.f; }

  __shared__ __attribute__((aligned(64))) __bf16 lds_p[16][32];

  const int nch = (qidx + 2) / 2;           // ceil((qidx+1)/2) 32-key chunks
  for (int ch = 0; ch < nch; ++ch) {
    const int kb0 = ch * 32;

    // prefetch next chunk's K rows into cache
    if (ch + 1 < nch) {
      __builtin_prefetch(Kp + (size_t)(kb0 + 32 + ln) * HDIM, 0, 1);
      __builtin_prefetch(Kp + (size_t)(kb0 + 48 + ln) * HDIM, 0, 1);
    }

    // ---- S = Q * K^T for 32 keys (two 16-key subtiles) ----
    const __bf16* Kl = Kp + (size_t)(kb0 + ln) * HDIM;
    v8f slo = {};
    slo = wmma_bf16(a0, load16(Kl + 16 * g), slo);
    slo = wmma_bf16(a1, load16(Kl + 32 + 16 * g), slo);
    const __bf16* Kh = Kp + (size_t)(kb0 + 16 + ln) * HDIM;
    v8f shi = {};
    shi = wmma_bf16(a0, load16(Kh + 16 * g), shi);
    shi = wmma_bf16(a1, load16(Kh + 32 + 16 * g), shi);

    // ---- hoist V B-tile loads: latency hidden under softmax VALU work ----
    v16bf vb[4];
#pragma unroll
    for (int j = 0; j < 4; ++j) {
      vb[j] = load16(Vp + (size_t)(16 * j + ln) * SEQ + kb0 + 16 * g);
    }

    // ---- online softmax (stats per C-layout row = r + 8*g) ----
    const int keylo = kb0 + ln;
    const int keyhi = kb0 + 16 + ln;
#pragma unroll
    for (int r = 0; r < 8; ++r) {
      const int row = q0 + r + 8 * g;
      float sl = slo[r] * 0.125f;           // 1/sqrt(64)
      float sh = shi[r] * 0.125f;
      if (keylo > row) sl = -1.0e30f;
      if (keyhi > row) sh = -1.0e30f;
      float tm = fmaxf(sl, sh);
#pragma unroll
      for (int msk = 1; msk < 16; msk <<= 1)
        tm = fmaxf(tm, __shfl_xor(tm, msk, 32));
      const float nm = fmaxf(mrow[r], tm);
      const float corr = __expf(mrow[r] - nm);
      const float pl = __expf(sl - nm);
      const float ph = __expf(sh - nm);
      float ps = pl + ph;
#pragma unroll
      for (int msk = 1; msk < 16; msk <<= 1)
        ps += __shfl_xor(ps, msk, 32);
      lrow[r] = lrow[r] * corr + ps;
      mrow[r] = nm;
      o[0][r] *= corr; o[1][r] *= corr; o[2][r] *= corr; o[3][r] *= corr;
      lds_p[r + 8 * g][ln] = (__bf16)pl;
      lds_p[r + 8 * g][16 + ln] = (__bf16)ph;
    }
    __syncthreads();
    // P tile back in A-matrix layout (16x32, local K = key-in-chunk)
    const v16bf pa = load16(&lds_p[ln][16 * g]);
    __syncthreads();

    // ---- O += P * V ----
#pragma unroll
    for (int j = 0; j < 4; ++j) o[j] = wmma_bf16(pa, vb[j], o[j]);
  }

  // ---- normalize and write to Of[B,T,C] ----
  const int b = bhid >> 4;
  const int h = bhid & 15;
#pragma unroll
  for (int r = 0; r < 8; ++r) {
    const float inv = 1.0f / lrow[r];
    const int t = q0 + r + 8 * g;
#pragma unroll
    for (int j = 0; j < 4; ++j) {
      Of[((size_t)b * SEQ + t) * CDIM + h * HDIM + 16 * j + ln] = o[j][r] * inv;
    }
  }
}

// ---------------------------------------------------------------------------
// Host launcher
// ---------------------------------------------------------------------------
extern "C" void kernel_launch(void* const* d_in, const int* in_sizes, int n_in,
                              void* d_out, int out_size, void* d_ws, size_t ws_size,
                              hipStream_t stream) {
  (void)in_sizes; (void)n_in; (void)out_size; (void)ws_size;
  const float* x      = (const float*)d_in[0];
  // d_in[1] = mask (causal tril; hard-coded in attn kernel)
  const float* w_attn = (const float*)d_in[2];
  const float* b_attn = (const float*)d_in[3];
  const float* w_proj = (const float*)d_in[4];
  const float* b_proj = (const float*)d_in[5];
  float* out = (float*)d_out;

  // ---- workspace carve-up ----
  char* p = (char*)d_ws;
  auto alloc = [&](size_t bytes) -> void* {
    void* r = (void*)p;
    p += (bytes + 255) & ~(size_t)255;
    return r;
  };
  __bf16* xb   = (__bf16*)alloc((size_t)MROWS * CDIM * 2);  // x in bf16
  __bf16* waT  = (__bf16*)alloc((size_t)NQKV * CDIM * 2);   // w_attn^T bf16
  __bf16* wpT  = (__bf16*)alloc((size_t)CDIM * CDIM * 2);   // w_proj^T bf16
  __bf16* qb   = (__bf16*)alloc((size_t)BH * SEQ * HDIM * 2);
  __bf16* kbuf = (__bf16*)alloc((size_t)BH * SEQ * HDIM * 2);
  __bf16* vT   = (__bf16*)alloc((size_t)BH * HDIM * SEQ * 2);
  float*  Of   = (float*)alloc((size_t)MROWS * CDIM * 4);   // attention output f32
  __bf16* ab   = (__bf16*)alloc((size_t)MROWS * CDIM * 2);  // attn out bf16
  float*  xsq  = (float*)alloc((size_t)MROWS * 4);
  float*  asq  = (float*)alloc((size_t)MROWS * 4);
  float*  wsqa = (float*)alloc((size_t)NQKV * 4);
  float*  wsqp = (float*)alloc((size_t)CDIM * 4);

  const float scale_qkv  = sqrtf((float)NQKV) / log1pf((float)NQKV);
  const float scale_proj = sqrtf((float)CDIM) / log1pf((float)CDIM);

  // 1) x -> bf16 + row sumsq
  rows_to_bf16_sumsq<<<MROWS, 256, 0, stream>>>(x, xb, xsq, CDIM);
  // 2) weight transposes + col sumsq (wsq accumulated via atomics)
  hipMemsetAsync(wsqa, 0, (size_t)NQKV * 4, stream);
  hipMemsetAsync(wsqp, 0, (size_t)CDIM * 4, stream);
  transpose_bf16_colsumsq<<<dim3(NQKV / 32, CDIM / 32), dim3(32, 8), 0, stream>>>(
      w_attn, waT, wsqa, CDIM, NQKV);
  transpose_bf16_colsumsq<<<dim3(CDIM / 32, CDIM / 32), dim3(32, 8), 0, stream>>>(
      w_proj, wpT, wsqp, CDIM, CDIM);
  // 3) YAT QKV GEMM (scatters q/k/v)
  yat_qkv_kernel<<<dim3(MROWS / 16, NQKV / 512), 256, 0, stream>>>(
      xb, waT, xsq, wsqa, b_attn, qb, kbuf, vT, scale_qkv);
  // 4) RoPE in place on q,k
  {
    const int total = BH * SEQ * (HDIM / 2);
    rope_kernel<<<(total + 255) / 256, 256, 0, stream>>>(qb, kbuf);
  }
  // 5) causal flash attention
  attn_kernel<<<BH * (SEQ / 16), 32, 0, stream>>>(qb, kbuf, vT, Of);
  // 6) attention output -> bf16 + row sumsq
  rows_to_bf16_sumsq<<<MROWS, 256, 0, stream>>>(Of, ab, asq, CDIM);
  // 7) YAT projection -> final f32 output
  yat_proj_kernel<<<dim3(MROWS / 16, CDIM / 512), 256, 0, stream>>>(
      ab, wpT, asq, wsqp, b_proj, out, scale_proj);
}
